// BiDAFAttention_11493332484256
// MI455X (gfx1250) — matrix operand order repaired and checked by reading, not compiled, not run on hardware
//
#include <hip/hip_runtime.h>
#include <hip/hip_bf16.h>
#include <cstdint>

typedef __attribute__((ext_vector_type(16))) _Float16 v16h;
typedef __attribute__((ext_vector_type(8)))  _Float16 half8;
typedef __attribute__((ext_vector_type(8)))  float    v8f;

constexpr int B  = 16;
constexpr int CL = 1024;
constexpr int QL = 128;
constexpr int H  = 512;
constexpr float NEGV = -1e30f;

// ---------------------------------------------------------------------------
// Prep kernels: f32 -> f16 casts, rank-1 projections, copy c into out seg 0.
// ---------------------------------------------------------------------------
__global__ void prep_c_kernel(const float* __restrict__ c,
                              const float* __restrict__ c_weight,
                              const float* __restrict__ cq_weight,
                              _Float16* __restrict__ cA16,
                              float* __restrict__ sim_c,
                              float* __restrict__ out) {
  const int row = blockIdx.x;                 // 0 .. B*CL-1
  const float* crow = c + (size_t)row * H;
  float* orow = out + (size_t)row * (4 * H);  // segment 0 = c
  float partial = 0.f;
  for (int h = threadIdx.x; h < H; h += blockDim.x) {
    float v = crow[h];
    cA16[(size_t)row * H + h] = (_Float16)(v * cq_weight[h]);
    orow[h] = v;
    partial += v * c_weight[h];
  }
  __shared__ float red[256];
  red[threadIdx.x] = partial;
  __syncthreads();
  for (int s = 128; s > 0; s >>= 1) {
    if ((int)threadIdx.x < s) red[threadIdx.x] += red[threadIdx.x + s];
    __syncthreads();
  }
  if (threadIdx.x == 0) sim_c[row] = red[0];
}

__global__ void prep_q_kernel(const float* __restrict__ q,
                              const float* __restrict__ q_weight,
                              _Float16* __restrict__ q16,
                              float* __restrict__ sim_q) {
  const int row = blockIdx.x;                 // 0 .. B*QL-1
  const float* qrow = q + (size_t)row * H;
  float partial = 0.f;
  for (int h = threadIdx.x; h < H; h += blockDim.x) {
    float v = qrow[h];
    q16[(size_t)row * H + h] = (_Float16)v;
    partial += v * q_weight[h];
  }
  __shared__ float red[256];
  red[threadIdx.x] = partial;
  __syncthreads();
  for (int s = 128; s > 0; s >>= 1) {
    if ((int)threadIdx.x < s) red[threadIdx.x] += red[threadIdx.x + s];
    __syncthreads();
  }
  if (threadIdx.x == 0) sim_q[row] = red[0];
}

// ---------------------------------------------------------------------------
// Batched LDS-tiled transpose: f32 [R x C] row-major -> f16 [C x R] row-major.
// grid = (R/32, C/32, B), block = (32, 8).
// ---------------------------------------------------------------------------
__global__ void transpose_f32_to_f16_kernel(const float* __restrict__ src,
                                            _Float16* __restrict__ dst,
                                            int R, int C) {
  __shared__ float tile[32][33];
  const int b  = blockIdx.z;
  const int r0 = blockIdx.x * 32;
  const int c0 = blockIdx.y * 32;
  const float* s = src + (size_t)b * R * C;
  _Float16*   d = dst + (size_t)b * R * C;
  const int tx = threadIdx.x;   // 0..31
  const int ty = threadIdx.y;   // 0..7
#pragma unroll
  for (int rr = ty; rr < 32; rr += 8)
    tile[rr][tx] = s[(size_t)(r0 + rr) * C + c0 + tx];
  __syncthreads();
#pragma unroll
  for (int rr = ty; rr < 32; rr += 8)
    d[(size_t)(c0 + rr) * R + r0 + tx] = (_Float16)tile[tx][rr];
}

// ---------------------------------------------------------------------------
// Fragment loaders (wave32, CDNA5 ISA 7.12.2 layouts). A is MxK row-major,
// Bt is NxK row-major (B transposed) -> all fragments load as b128 pairs.
// ---------------------------------------------------------------------------
__device__ __forceinline__ v16h load_a_frag(const _Float16* __restrict__ A, int lda,
                                            int mrow, int k0, int kb) {
  const _Float16* ap = A + (size_t)mrow * lda + k0 + kb;
  half8 lo = *(const half8*)(ap);
  half8 hi = *(const half8*)(ap + 16);
  v16h v;
#pragma unroll
  for (int h = 0; h < 8; ++h) { v[h] = lo[h]; v[h + 8] = hi[h]; }
  return v;
}

__device__ __forceinline__ v16h load_bt_frag(const _Float16* __restrict__ Bt, int ldbt,
                                             int ncol, int k0, int kc) {
  const _Float16* bp = Bt + (size_t)ncol * ldbt + k0 + kc;
  half8 lo = *(const half8*)(bp);
  half8 hi = *(const half8*)(bp + 16);
  v16h v;
#pragma unroll
  for (int h = 0; h < 8; ++h) { v[h] = lo[h]; v[h + 8] = hi[h]; }
  return v;
}

// 32x32 macro-tile accumulate: 4 v_wmma per K-step of 32.
__device__ __forceinline__ void wmma_tile32(const _Float16* __restrict__ A, int lda,
                                            const _Float16* __restrict__ Bt, int ldbt,
                                            int m0, int n0, int K, int lane,
                                            v8f acc[2][2]) {
  const int mr = lane & 15;
  const int kb = (lane & 16) ? 8  : 0;
  const int kc = (lane & 16) ? 8  : 0;
  for (int k0 = 0; k0 < K; k0 += 32) {
    v16h av[2], bv[2];
#pragma unroll
    for (int t = 0; t < 2; ++t) {
      av[t] = load_a_frag(A, lda, m0 + t * 16 + mr, k0, kb);
      bv[t] = load_bt_frag(Bt, ldbt, n0 + t * 16 + mr, k0, kc);
    }
#pragma unroll
    for (int i = 0; i < 2; ++i)
#pragma unroll
      for (int j = 0; j < 2; ++j)
        acc[i][j] = __builtin_amdgcn_wmma_f32_16x16x32_f16(false, av[i], false, bv[j],
                                                           (short)0, acc[i][j], false, false);
  }
}

// ---------------------------------------------------------------------------
// S[b,i,j] = sim_c[b,i] + sim_q[b,j] + bias + (c*cqw)[b,i,:] . q[b,j,:]
// A = cA16 [CL x H], Bt = q16 [QL x H] (N x K).
// ---------------------------------------------------------------------------
__global__ void gemm_s_kernel(const _Float16* __restrict__ cA16,
                              const _Float16* __restrict__ q16,
                              const float* __restrict__ sim_c,
                              const float* __restrict__ sim_q,
                              const float* __restrict__ bias,
                              float* __restrict__ S) {
  const int lane = threadIdx.x & 31;
  const int wave = threadIdx.x >> 5;
  const int tile = blockIdx.x * (blockDim.x >> 5) + wave;
  const int tpb  = (CL / 32) * (QL / 32);
  const int b  = tile / tpb;
  const int t  = tile % tpb;
  const int m0 = (t / (QL / 32)) * 32;
  const int n0 = (t % (QL / 32)) * 32;
  const _Float16* A  = cA16 + (size_t)b * CL * H;
  const _Float16* Bt = q16  + (size_t)b * QL * H;
  v8f acc[2][2] = {};
  wmma_tile32(A, H, Bt, H, m0, n0, H, lane, acc);
  const float bb = bias[0];
  const int mb = (lane & 16) ? 8 : 0;
#pragma unroll
  for (int i = 0; i < 2; ++i) {
#pragma unroll
    for (int j = 0; j < 2; ++j) {
      const int n = n0 + j * 16 + (lane & 15);
      const float sq = sim_q[b * QL + n] + bb;
#pragma unroll
      for (int r = 0; r < 8; ++r) {
        int m = m0 + i * 16 + mb + r;
        S[((size_t)b * CL + m) * QL + n] = acc[i][j][r] + sim_c[b * CL + m] + sq;
      }
    }
  }
}

// ---------------------------------------------------------------------------
// Softmax over j (QL) with q_mask  -> s1 (f16, [B*CL, QL])
// ---------------------------------------------------------------------------
__global__ void softmax_rows_kernel(const float* __restrict__ S,
                                    const int* __restrict__ q_mask,
                                    _Float16* __restrict__ s1) {
  const int row = blockIdx.x;      // b*CL + i
  const int b = row / CL;
  const int j = threadIdx.x;       // blockDim == QL
  float m = (float)q_mask[b * QL + j];
  float v = m * S[(size_t)row * QL + j] + (1.f - m) * NEGV;
  __shared__ float red[QL];
  red[j] = v; __syncthreads();
  for (int s = QL / 2; s > 0; s >>= 1) {
    if (j < s) red[j] = fmaxf(red[j], red[j + s]);
    __syncthreads();
  }
  float mx = red[0]; __syncthreads();
  float e = __expf(v - mx);
  red[j] = e; __syncthreads();
  for (int s = QL / 2; s > 0; s >>= 1) {
    if (j < s) red[j] += red[j + s];
    __syncthreads();
  }
  s1[(size_t)row * QL + j] = (_Float16)(e / red[0]);
}

// ---------------------------------------------------------------------------
// Softmax over i (CL) with c_mask -> s2T (f16, stored transposed [B*QL, CL])
// ---------------------------------------------------------------------------
__global__ void softmax_cols_kernel(const float* __restrict__ S,
                                    const int* __restrict__ c_mask,
                                    _Float16* __restrict__ s2T) {
  const int col = blockIdx.x;      // b*QL + j
  const int b = col / QL;
  const int j = col % QL;
  const float* Sb = S + (size_t)b * CL * QL;
  const int*  mb = c_mask + b * CL;
  constexpr int R = CL / 256;
  float x[R];
  float lm = -3.4e38f;
#pragma unroll
  for (int r = 0; r < R; ++r) {
    int i = threadIdx.x + r * 256;
    float m = (float)mb[i];
    float v = m * Sb[(size_t)i * QL + j] + (1.f - m) * NEGV;
    x[r] = v;
    lm = fmaxf(lm, v);
  }
  __shared__ float red[256];
  red[threadIdx.x] = lm; __syncthreads();
  for (int s = 128; s > 0; s >>= 1) {
    if ((int)threadIdx.x < s) red[threadIdx.x] = fmaxf(red[threadIdx.x], red[threadIdx.x + s]);
    __syncthreads();
  }
  float mx = red[0]; __syncthreads();
  float ls = 0.f;
#pragma unroll
  for (int r = 0; r < R; ++r) { x[r] = __expf(x[r] - mx); ls += x[r]; }
  red[threadIdx.x] = ls; __syncthreads();
  for (int s = 128; s > 0; s >>= 1) {
    if ((int)threadIdx.x < s) red[threadIdx.x] += red[threadIdx.x + s];
    __syncthreads();
  }
  float inv = 1.f / red[0];
  _Float16* drow = s2T + (size_t)col * CL;
#pragma unroll
  for (int r = 0; r < R; ++r) {
    int i = threadIdx.x + r * 256;
    drow[i] = (_Float16)(x[r] * inv);
  }
}

// ---------------------------------------------------------------------------
// t = s2^T @ c : A = s2T [QL x CL], Bt = cT16 [H x CL] (N x K).
// Stores result transposed: tT16 [H x QL] (so gemm_ab gets an N x K operand).
// ---------------------------------------------------------------------------
__global__ void gemm_t_kernel(const _Float16* __restrict__ s2T,
                              const _Float16* __restrict__ cT16,
                              _Float16* __restrict__ tT16) {
  const int lane = threadIdx.x & 31;
  const int wave = threadIdx.x >> 5;
  const int tile = blockIdx.x * (blockDim.x >> 5) + wave;
  const int tpb  = (QL / 32) * (H / 32);
  const int b  = tile / tpb;
  const int t  = tile % tpb;
  const int m0 = (t / (H / 32)) * 32;
  const int n0 = (t % (H / 32)) * 32;
  const _Float16* A  = s2T  + (size_t)b * QL * CL;
  const _Float16* Bt = cT16 + (size_t)b * H * CL;
  v8f acc[2][2] = {};
  wmma_tile32(A, CL, Bt, CL, m0, n0, CL, lane, acc);
  const int mb = (lane & 16) ? 8 : 0;
#pragma unroll
  for (int i = 0; i < 2; ++i) {
#pragma unroll
    for (int j = 0; j < 2; ++j) {
      const int n = n0 + j * 16 + (lane & 15);
      half8 v;
#pragma unroll
      for (int r = 0; r < 8; ++r) v[r] = (_Float16)acc[i][j][r];
      // transposed store: row n (length QL), 8 contiguous halfs at m
      *(half8*)(tT16 + ((size_t)b * H + n) * QL + m0 + i * 16 + mb) = v;
    }
  }
}

// ---------------------------------------------------------------------------
// Fused: a = s1 @ q and bmat = s1 @ t, sharing the s1 A-fragments.
// A = s1 [CL x QL]; BtQ = qT16 [H x QL]; BtT = tT16 [H x QL].
// Per K-step: 6 fragments (12 b128 loads) feed 8 v_wmma.
// Epilogue: out seg1 = a, seg2 = c*a, seg3 = c*b (single c read).
// ---------------------------------------------------------------------------
__global__ void gemm_ab_kernel(const _Float16* __restrict__ s1,
                               const _Float16* __restrict__ qT16,
                               const _Float16* __restrict__ tT16,
                               const float* __restrict__ c,
                               float* __restrict__ out) {
  const int lane = threadIdx.x & 31;
  const int wave = threadIdx.x >> 5;
  const int tile = blockIdx.x * (blockDim.x >> 5) + wave;
  const int tpb  = (CL / 32) * (H / 32);
  const int b  = tile / tpb;
  const int t  = tile % tpb;
  const int m0 = (t / (H / 32)) * 32;
  const int n0 = (t % (H / 32)) * 32;
  const _Float16* A   = s1   + (size_t)b * CL * QL;
  const _Float16* BtQ = qT16 + (size_t)b * H * QL;
  const _Float16* BtT = tT16 + (size_t)b * H * QL;
  const int mr = lane & 15;
  const int kb = (lane & 16) ? 8  : 0;
  const int kc = (lane & 16) ? 8  : 0;
  v8f accA[2][2] = {};
  v8f accB[2][2] = {};
  for (int k0 = 0; k0 < QL; k0 += 32) {
    v16h av[2], bq[2], bt[2];
#pragma unroll
    for (int u = 0; u < 2; ++u) {
      av[u] = load_a_frag(A, QL, m0 + u * 16 + mr, k0, kb);
      bq[u] = load_bt_frag(BtQ, QL, n0 + u * 16 + mr, k0, kc);
      bt[u] = load_bt_frag(BtT, QL, n0 + u * 16 + mr, k0, kc);
    }
#pragma unroll
    for (int i = 0; i < 2; ++i) {
#pragma unroll
      for (int j = 0; j < 2; ++j) {
        accA[i][j] = __builtin_amdgcn_wmma_f32_16x16x32_f16(false, av[i], false, bq[j],
                                                            (short)0, accA[i][j], false, false);
        accB[i][j] = __builtin_amdgcn_wmma_f32_16x16x32_f16(false, av[i], false, bt[j],
                                                            (short)0, accB[i][j], false, false);
      }
    }
  }
  const int mb = (lane & 16) ? 8 : 0;
#pragma unroll
  for (int i = 0; i < 2; ++i) {
#pragma unroll
    for (int j = 0; j < 2; ++j) {
      const int n = n0 + j * 16 + (lane & 15);
#pragma unroll
      for (int r = 0; r < 8; ++r) {
        int m = m0 + i * 16 + mb + r;
        size_t row = (size_t)b * CL + m;
        float av = accA[i][j][r];
        float bv = accB[i][j][r];
        float cv = c[row * H + n];
        float* orow = out + row * (4 * H);
        orow[H + n]     = av;
        orow[2 * H + n] = cv * av;
        orow[3 * H + n] = cv * bv;
      }
    }
  }
}

// ---------------------------------------------------------------------------
extern "C" void kernel_launch(void* const* d_in, const int* in_sizes, int n_in,
                              void* d_out, int out_size, void* d_ws, size_t ws_size,
                              hipStream_t stream) {
  const float* c         = (const float*)d_in[0];
  const float* q         = (const float*)d_in[1];
  const int*   c_mask    = (const int*)  d_in[2];
  const int*   q_mask    = (const int*)  d_in[3];
  const float* c_weight  = (const float*)d_in[4];
  const float* q_weight  = (const float*)d_in[5];
  const float* cq_weight = (const float*)d_in[6];
  const float* bias      = (const float*)d_in[7];
  float* out = (float*)d_out;

  char* w = (char*)d_ws;
  size_t off = 0;
  auto take = [&](size_t bytes) -> char* {
    char* p = w + off;
    off += (bytes + 255) & ~(size_t)255;
    return p;
  };
  _Float16* cA16  = (_Float16*)take((size_t)B * CL * H * 2);   // (c*cqw) f16  [B][CL][H]
  _Float16* cT16  = (_Float16*)take((size_t)B * H * CL * 2);   // c^T f16      [B][H][CL]
  _Float16* q16   = (_Float16*)take((size_t)B * QL * H * 2);   // q f16        [B][QL][H]
  _Float16* qT16  = (_Float16*)take((size_t)B * H * QL * 2);   // q^T f16      [B][H][QL]
  _Float16* s1    = (_Float16*)take((size_t)B * CL * QL * 2);  // softmax_j    [B][CL][QL]
  _Float16* s2T   = (_Float16*)take((size_t)B * QL * CL * 2);  // softmax_i^T  [B][QL][CL]
  _Float16* tT16  = (_Float16*)take((size_t)B * H * QL * 2);   // (s2^T c)^T   [B][H][QL]
  float*    S     = (float*)   take((size_t)B * CL * QL * 4);  // raw scores
  float*    sim_c = (float*)   take((size_t)B * CL * 4);
  float*    sim_q = (float*)   take((size_t)B * QL * 4);

  prep_c_kernel<<<B * CL, 256, 0, stream>>>(c, c_weight, cq_weight, cA16, sim_c, out);
  prep_q_kernel<<<B * QL, 256, 0, stream>>>(q, q_weight, q16, sim_q);
  transpose_f32_to_f16_kernel<<<dim3(CL / 32, H / 32, B), dim3(32, 8), 0, stream>>>(c, cT16, CL, H);
  transpose_f32_to_f16_kernel<<<dim3(QL / 32, H / 32, B), dim3(32, 8), 0, stream>>>(q, qT16, QL, H);

  {
    int tiles = B * (CL / 32) * (QL / 32);        // 2048 macro-tiles
    gemm_s_kernel<<<tiles / 4, 128, 0, stream>>>(cA16, q16, sim_c, sim_q, bias, S);
  }

  softmax_rows_kernel<<<B * CL, QL, 0, stream>>>(S, q_mask, s1);
  softmax_cols_kernel<<<B * QL, 256, 0, stream>>>(S, c_mask, s2T);

  {
    int tiles = B * (QL / 32) * (H / 32);         // 1024 macro-tiles
    gemm_t_kernel<<<tiles / 4, 128, 0, stream>>>(s2T, cT16, tT16);
  }
  {
    int tiles = B * (CL / 32) * (H / 32);         // 8192 macro-tiles
    gemm_ab_kernel<<<tiles / 4, 128, 0, stream>>>(s1, qT16, tT16, c, out);
  }
}
